// DenseAE_29360396435969
// MI455X (gfx1250) — matrix operand order repaired and checked by reading, not compiled, not run on hardware
//
#include <hip/hip_runtime.h>
#include <hip/hip_bf16.h>
#include <stdint.h>

typedef __attribute__((ext_vector_type(16))) __bf16    v16bf;
typedef __attribute__((ext_vector_type(8)))  float     v8f;
typedef __attribute__((ext_vector_type(8)))  uint32_t  v8u;
typedef __attribute__((__vector_size__(16))) int       i32x4;   // matches builtin param type

#define BMM 256   // block rows
#define BNN 128   // block cols
#define BK  32    // K per stage (== WMMA K)
#define LDSK 40   // BK + 8 bf16 pad: 80B row stride, 16B aligned, conflict-free banks

// ---------- CDNA5 async global->LDS copy (ASYNCcnt path), 16B granule ----------
__device__ __forceinline__ void async_copy_b128(const __hip_bfloat16* gsrc, unsigned short* ldst)
{
#if __has_builtin(__builtin_amdgcn_global_load_async_to_lds_b128)
    void* g = (void*)gsrc;   // strip const; builtin takes non-const global ptr
    void* l = (void*)ldst;
    __builtin_amdgcn_global_load_async_to_lds_b128(
        (__attribute__((address_space(1))) i32x4*)g,
        (__attribute__((address_space(3))) i32x4*)l, 0, 0);
#else
    unsigned int loff = (unsigned int)(unsigned long long)(uintptr_t)ldst; // flat LDS addr: low 32 bits = LDS offset
    asm volatile("global_load_async_to_lds_b128 %0, %1, off"
                 :: "v"(loff), "v"((unsigned long long)(uintptr_t)gsrc)
                 : "memory");
#endif
}

__device__ __forceinline__ void wait_async0()
{
#if __has_builtin(__builtin_amdgcn_s_wait_asynccnt)
    __builtin_amdgcn_s_wait_asynccnt(0);
#else
    asm volatile("s_wait_asynccnt 0x0" ::: "memory");
#endif
}

// ---------------- f32 -> bf16 elementwise (4 elems/thread) ----------------
__global__ __launch_bounds__(256)
void f32_to_bf16_kernel(const float* __restrict__ src, __hip_bfloat16* __restrict__ dst, long n)
{
    long i = ((long)blockIdx.x * 256 + threadIdx.x) * 4;
    if (i + 3 >= n) return;
    float4 v = *(const float4*)(src + i);
    dst[i + 0] = __float2bfloat16(v.x);
    dst[i + 1] = __float2bfloat16(v.y);
    dst[i + 2] = __float2bfloat16(v.z);
    dst[i + 3] = __float2bfloat16(v.w);
}

// ---------------- f32 KxN -> bf16 NxK transpose (32x32 LDS tiles) ----------------
__global__ __launch_bounds__(256)
void transpose_f32_to_bf16(const float* __restrict__ src, __hip_bfloat16* __restrict__ dst,
                           int K, int N)
{
    __shared__ float tile[32][33];
    const int n0 = blockIdx.x * 32;
    const int k0 = blockIdx.y * 32;
    const int tx = threadIdx.x & 31;
    const int ty = threadIdx.x >> 5;
    #pragma unroll
    for (int r = ty; r < 32; r += 8)
        tile[r][tx] = src[(long)(k0 + r) * N + n0 + tx];
    __syncthreads();
    #pragma unroll
    for (int r = ty; r < 32; r += 8)
        dst[(long)(n0 + r) * K + k0 + tx] = __float2bfloat16(tile[tx][r]);
}

// ---------------- bf16 WMMA GEMM:  C = epi(A[MxK] * Bt[NxK]^T + bias) ----------------
// EPI: 0 = ReLU -> bf16 out, 1 = identity -> f32 out, 2 = sigmoid -> f32 out
// 256 threads = 8 waves, block tile 256x128, wave tile 64x64 (16 WMMAs / K-step),
// double-buffered LDS fed by async global->LDS copies.
template<int EPI>
__global__ __launch_bounds__(256)
void gemm_bf16_wmma(const __hip_bfloat16* __restrict__ A,
                    const __hip_bfloat16* __restrict__ Bt,
                    const float* __restrict__ bias,
                    void* __restrict__ Cout,
                    int M, int N, int K)
{
    __shared__ __align__(16) unsigned short sA[2 * BMM * LDSK];  // 40960 B
    __shared__ __align__(16) unsigned short sB[2 * BNN * LDSK];  // 20480 B

    const int t    = threadIdx.x;
    const int m0   = blockIdx.y * BMM;
    const int n0   = blockIdx.x * BNN;
    const int wave = t >> 5;
    const int lane = t & 31;
    const int wm   = wave & 3;        // 4 waves along M (64 rows each)
    const int wn   = wave >> 2;       // 2 waves along N (64 cols each)
    const int half = lane >> 4;       // K-half selector by lane group
    const int l16  = lane & 15;

    v8f acc[4][4];
    #pragma unroll
    for (int i = 0; i < 4; ++i)
        #pragma unroll
        for (int j = 0; j < 4; ++j)
            #pragma unroll
            for (int r = 0; r < 8; ++r) acc[i][j][r] = 0.0f;

    // stage one 256x32 A tile + 128x32 B tile into buffer `buf` (async, no wait)
    auto stage = [&](int buf, int kt) {
        unsigned short* dA = sA + buf * (BMM * LDSK);
        unsigned short* dB = sB + buf * (BNN * LDSK);
        #pragma unroll
        for (int u = 0; u < 4; ++u) {            // A: 1024 b128 chunks
            int c = t + u * 256;
            int row = c >> 2, kc = (c & 3) << 3;
            async_copy_b128(A + (long)(m0 + row) * K + kt + kc, dA + row * LDSK + kc);
        }
        #pragma unroll
        for (int u = 0; u < 2; ++u) {            // B: 512 b128 chunks
            int c = t + u * 256;
            int row = c >> 2, kc = (c & 3) << 3;
            async_copy_b128(Bt + (long)(n0 + row) * K + kt + kc, dB + row * LDSK + kc);
        }
    };

    stage(0, 0);
    wait_async0();
    __syncthreads();

    const int nkt = K / BK;
    for (int it = 0; it < nkt; ++it) {
        const int buf = it & 1;
        if (it + 1 < nkt) stage(buf ^ 1, (it + 1) * BK);   // async prefetch next tile

        const unsigned short* cA = sA + buf * (BMM * LDSK);
        const unsigned short* cB = sB + buf * (BNN * LDSK);

        // B fragments: col = l16; VGPR0-3: K = half*8+0..7, VGPR4-7: K = 16+half*8+0..7
        v8u bfrag[4];
        #pragma unroll
        for (int j = 0; j < 4; ++j) {
            const uint4* p = (const uint4*)(cB + (wn * 64 + j * 16 + l16) * LDSK + half * 8);
            uint4 lo = p[0], hi = p[2];
            bfrag[j][0] = lo.x; bfrag[j][1] = lo.y; bfrag[j][2] = lo.z; bfrag[j][3] = lo.w;
            bfrag[j][4] = hi.x; bfrag[j][5] = hi.y; bfrag[j][6] = hi.z; bfrag[j][7] = hi.w;
        }
        // A fragments: row = l16; VGPR0-3: K = half*8+0..7, VGPR4-7: K = 16+half*8+0..7
        #pragma unroll
        for (int i = 0; i < 4; ++i) {
            const unsigned short* ab = cA + (wm * 64 + i * 16 + l16) * LDSK + half * 8;
            uint4 lo = *(const uint4*)(ab);
            uint4 hi = *(const uint4*)(ab + 16);
            v8u af;
            af[0] = lo.x; af[1] = lo.y; af[2] = lo.z; af[3] = lo.w;
            af[4] = hi.x; af[5] = hi.y; af[6] = hi.z; af[7] = hi.w;
            v16bf a = __builtin_bit_cast(v16bf, af);
            #pragma unroll
            for (int j = 0; j < 4; ++j) {
                v16bf b = __builtin_bit_cast(v16bf, bfrag[j]);
                acc[i][j] = __builtin_amdgcn_wmma_f32_16x16x32_bf16(
                    false, a, false, b, (short)0, acc[i][j], false, false);
            }
        }

        wait_async0();       // my async writes into buf^1 have landed
        __syncthreads();     // everyone's have; also fences reuse of `buf` next iter
    }

    // epilogue: C VGPR r -> M = r (lanes 0-15) / M = r+8 (lanes 16-31), N = l16
    #pragma unroll
    for (int i = 0; i < 4; ++i) {
        #pragma unroll
        for (int j = 0; j < 4; ++j) {
            int col   = n0 + wn * 64 + j * 16 + l16;
            float bv  = bias[col];
            int rbase = m0 + wm * 64 + i * 16 + half * 8;
            #pragma unroll
            for (int r = 0; r < 8; ++r) {
                float v  = acc[i][j][r] + bv;
                long idx = (long)(rbase + r) * N + col;
                if (EPI == 0) {
                    v = v > 0.0f ? v : 0.0f;
                    ((__hip_bfloat16*)Cout)[idx] = __float2bfloat16(v);
                } else if (EPI == 1) {
                    ((float*)Cout)[idx] = v;
                } else {
                    ((float*)Cout)[idx] = 1.0f / (1.0f + __expf(-v));
                }
            }
        }
    }
}

// ---------------- k-sparse: exact top-k threshold per row via MSB radix select ----------------
__global__ __launch_bounds__(256)
void ksparse_kernel(const float* __restrict__ H2, __hip_bfloat16* __restrict__ Hs,
                    int N, const int* __restrict__ nbp)
{
    const int  row = blockIdx.x;
    const int  kk  = *nbp;
    const float* h = H2 + (long)row * N;

    float    val[8];
    uint32_t key[8];
    #pragma unroll
    for (int i = 0; i < 8; ++i) {
        int idx = threadIdx.x + i * 256;
        float x = h[idx];
        val[i] = x;
        uint32_t b = __float_as_uint(x);
        key[i] = b ^ ((b >> 31) ? 0xFFFFFFFFu : 0x80000000u);   // order-preserving map
    }

    __shared__ uint32_t hist[16];
    uint32_t prefix = 0;
    uint32_t krem   = (uint32_t)kk;

    for (int d = 28; d >= 0; d -= 4) {
        uint32_t mask = (d == 28) ? 0u : (0xFFFFFFFFu << (d + 4));
        if (threadIdx.x < 16) hist[threadIdx.x] = 0;
        __syncthreads();
        #pragma unroll
        for (int i = 0; i < 8; ++i)
            if ((key[i] & mask) == prefix)
                atomicAdd(&hist[(key[i] >> d) & 15], 1u);
        __syncthreads();
        uint32_t cum = 0; int sel = 0; uint32_t above = 0;
        #pragma unroll
        for (int b = 15; b >= 0; --b) {
            uint32_t c = hist[b];
            if (cum + c >= krem) { sel = b; above = cum; break; }
            cum += c;
        }
        prefix |= (uint32_t)sel << d;
        krem   -= above;
        __syncthreads();
    }

    uint32_t tb = (prefix & 0x80000000u) ? (prefix ^ 0x80000000u) : ~prefix;
    float thresh = __uint_as_float(tb);

    #pragma unroll
    for (int i = 0; i < 8; ++i) {
        int idx = threadIdx.x + i * 256;
        float o = (val[i] >= thresh) ? val[i] : 0.0f;   // ties pass, matches reference
        Hs[(long)row * N + idx] = __float2bfloat16(o);
    }
}

// ---------------- host launch ----------------
extern "C" void kernel_launch(void* const* d_in, const int* in_sizes, int n_in,
                              void* d_out, int out_size, void* d_ws, size_t ws_size,
                              hipStream_t stream)
{
    (void)in_sizes; (void)n_in; (void)out_size; (void)ws_size;
    const float* X  = (const float*)d_in[0];
    const float* W1 = (const float*)d_in[1];
    const float* b1 = (const float*)d_in[2];
    const float* W2 = (const float*)d_in[3];
    const float* b2 = (const float*)d_in[4];
    const float* W3 = (const float*)d_in[5];
    const float* b3 = (const float*)d_in[6];
    const int*  nbp = (const int*)d_in[7];

    const int BATCH = 8192, DIN = 64 * 64 * 3, HID = 2048;

    // workspace carve-out (~424 MB total)
    char* ws = (char*)d_ws;
    __hip_bfloat16* Xb  = (__hip_bfloat16*)ws;  ws += (size_t)BATCH * DIN * 2;
    __hip_bfloat16* W1t = (__hip_bfloat16*)ws;  ws += (size_t)DIN * HID * 2;
    __hip_bfloat16* W2t = (__hip_bfloat16*)ws;  ws += (size_t)HID * HID * 2;
    __hip_bfloat16* W3t = (__hip_bfloat16*)ws;  ws += (size_t)HID * DIN * 2;
    __hip_bfloat16* H1  = (__hip_bfloat16*)ws;  ws += (size_t)BATCH * HID * 2;
    float*          H2  = (float*)ws;           ws += (size_t)BATCH * HID * 4;
    __hip_bfloat16* Hs  = (__hip_bfloat16*)ws;

    long nX = (long)BATCH * DIN;
    f32_to_bf16_kernel<<<(unsigned)(nX / 1024), 256, 0, stream>>>(X, Xb, nX);

    transpose_f32_to_bf16<<<dim3(HID / 32, DIN / 32), 256, 0, stream>>>(W1, W1t, DIN, HID);
    transpose_f32_to_bf16<<<dim3(HID / 32, HID / 32), 256, 0, stream>>>(W2, W2t, HID, HID);
    transpose_f32_to_bf16<<<dim3(DIN / 32, HID / 32), 256, 0, stream>>>(W3, W3t, HID, DIN);

    gemm_bf16_wmma<0><<<dim3(HID / BNN, BATCH / BMM), 256, 0, stream>>>(Xb, W1t, b1, H1, BATCH, HID, DIN);
    gemm_bf16_wmma<1><<<dim3(HID / BNN, BATCH / BMM), 256, 0, stream>>>(H1, W2t, b2, H2, BATCH, HID, HID);
    ksparse_kernel<<<BATCH, 256, 0, stream>>>(H2, Hs, HID, nbp);
    gemm_bf16_wmma<2><<<dim3(DIN / BNN, BATCH / BMM), 256, 0, stream>>>(Hs, W3t, b3, d_out, BATCH, DIN, HID);
}
